// Mimic_40827959116324
// MI455X (gfx1250) — matrix operand order repaired and checked by reading, not compiled, not run on hardware
//
#include <hip/hip_runtime.h>

// Compute-bound per-feature MLP (B=32768, F=64, H=64, L=2) on gfx1250.
// AI ~2100 FLOP/byte -> hidden GEMMs on v_wmma_f32_16x16x32_bf16 (f32 acc).
// TRANSPOSED formulation: h_new^T = W * h^T, so A = weights, B = activations^T.
//  * layer0 -> layer1: B fragment computed directly in registers (no LDS)
//  * layer1 epilogue: lane's 8 D values contiguous in k_out -> packed CELU
//    + one ds_store_b128 per 16x16 tile (bf16 [b][k] staging)
//  * layer2 epilogue: output dot folded in (packed fma), one __shfl_xor
//  * C fragment = inline 0; bias folded into epilogues as packed adds
//  * no in-loop barriers: staging is wave-private, same-wave LDS is in-order
//  * rolled tile loop + per-iteration memory clobber keeps VGPRs < 256
//    (no s_set_vgpr_msb traffic, high occupancy)
//  * NEW: branchless CELU  celu(x) = max(x, exp(min(x,0)) - 1)
//    - exact for x>0 (exp(0)-1 == 0), and e^x >= 1+x makes max() pick the
//      expm1 branch for x<=0. Removes all v_cmp->SGPR + v_cndmask chains
//      (and their s_delay_alu stalls); min/max are VOPD dual-issuable.
//
// Fragment layouts (wave32; C/D per CDNA5 ISA 7.12.2, A/B k order chosen here):
//   A (16x32 bf16): lane&15 = M; slot i <-> K = 16*hf + i   (per 32-wide chunk; this code's order)
//   B (32x16 bf16): lane&15 = N; slot i <-> K = 16*hf + i   (per 32-wide chunk; this code's order)
//   C/D (16x16 f32): lane&15 = N; VGPR r <-> M = r + 8*hf

namespace {

constexpr int F = 64;
constexpr int H = 64;
constexpr int L = 2;
constexpr int WAVES = 8;
constexpr int TILE_M = 16;
constexpr int BLOCK_ROWS = WAVES * TILE_M;                     // 128
constexpr int TILES_PER_BLOCK = 8;
constexpr int BATCH_PER_BLOCK = BLOCK_ROWS * TILES_PER_BLOCK;  // 1024
constexpr int HP = H + 8;   // bf16 row stride (144 B, keeps 16B-aligned chunks)

typedef __attribute__((ext_vector_type(16))) __bf16 v16bf;
typedef __attribute__((ext_vector_type(8)))  __bf16 v8bf;
typedef __attribute__((ext_vector_type(8)))  float  v8f;

__device__ __forceinline__ float fast_celu(float x) {
    // branchless CELU(alpha=1): max(x, exp(min(x,0)) - 1)
    return fmaxf(x, __expf(fminf(x, 0.0f)) - 1.0f);
}

__device__ __forceinline__ v16bf cat8(v8bf lo, v8bf hi) {
    return __builtin_shufflevector(lo, hi, 0, 1, 2, 3, 4, 5, 6, 7,
                                           8, 9, 10, 11, 12, 13, 14, 15);
}

__global__ __launch_bounds__(WAVES * 32) void
feature_mlp_kernel(const float* __restrict__ x,
                   const float* __restrict__ w_in,
                   const float* __restrict__ b_in,
                   const float* __restrict__ w_hid,
                   const float* __restrict__ b_hid,
                   const float* __restrict__ w_out,
                   const float* __restrict__ b_out,
                   float* __restrict__ out,
                   int Btot)
{
    __shared__ __align__(16) __bf16 wq[L][H][H];             // 16 KB bf16 W[l][kout][kin]
    __shared__ __align__(16) __bf16 hbuf[WAVES][TILE_M][HP]; // 18 KB h1^T as [b][k]
    __shared__ float s_win[H], s_bin[H];
    __shared__ float s_bh[L][H];
    __shared__ float s_wo[H];
    __shared__ float s_bo;

    const int f    = blockIdx.x;                 // feature (independent MLP)
    const int tid  = threadIdx.x;
    const int wave = tid >> 5;
    const int lane = tid & 31;
    const int mrow = lane & 15;                  // A: M index / B,D: batch column
    const int hf   = lane >> 4;                  // lane half

    // ---- stage this feature's weights into LDS ----
    for (int i = tid; i < L * H * H; i += blockDim.x) {
        const int l = i / (H * H);
        const int r = i - l * H * H;
        (&wq[0][0][0])[i] = (__bf16)w_hid[((size_t)l * F + f) * H * H + r];
    }
    for (int i = tid; i < H; i += blockDim.x) {
        s_win[i] = w_in[f * H + i];
        s_bin[i] = b_in[f * H + i];
        s_wo[i]  = w_out[f * H + i];
        for (int l = 0; l < L; ++l) s_bh[l][i] = b_hid[(l * F + f) * H + i];
    }
    if (tid == 0) s_bo = b_out[f];
    __syncthreads();   // only barrier: weights become block-visible

    const int block_base = blockIdx.y * BATCH_PER_BLOCK;

#pragma clang loop unroll(disable)
    for (int t = 0; t < TILES_PER_BLOCK; ++t) {
        // keep VGPR pressure bounded: re-read LDS invariants every tile
        asm volatile("" ::: "memory");

        const int row0 = block_base + t * BLOCK_ROWS + wave * TILE_M;
        const int rrow = min(row0 + mrow, Btot - 1);

        if (t + 1 < TILES_PER_BLOCK) {           // global_prefetch next tile's x
            const int prow = min(row0 + BLOCK_ROWS + mrow, Btot - 1);
            __builtin_prefetch(&x[(size_t)prow * F + f], 0, 0);
        }

        // ---- layer 0: CELU(x*w+b) packed straight into B fragments of h0^T
        //      (lane = batch column b = mrow; slot i <-> k = 32c + 16*hf + i)
        const float xv = x[(size_t)rrow * F + f];
        v16bf h0b[2];
#pragma unroll
        for (int c = 0; c < 2; ++c)
#pragma unroll
            for (int i = 0; i < 16; ++i) {
                const int k = 32 * c + 16 * hf + i;     // contiguous -> packed
                h0b[c][i] = (__bf16)fast_celu(xv * s_win[k] + s_bin[k]);
            }

        // ---- layer 1: h1^T = CELU(W1 * h0^T + b1), staged bf16 [b][k] ----
#pragma unroll
        for (int mt = 0; mt < 4; ++mt) {
            const v8bf* wrow = (const v8bf*)(&wq[0][mt * 16 + mrow][0]);
            const v16bf a0 = cat8(wrow[2 * hf],     wrow[2 * hf + 1]);   // K = 0..31
            const v16bf a1 = cat8(wrow[4 + 2 * hf], wrow[5 + 2 * hf]);   // K = 32..63

            v8f acc = {};   // inline SRC2 = 0
            acc = __builtin_amdgcn_wmma_f32_16x16x32_bf16(
                      false, a0, false, h0b[0], (short)0, acc, false, false);
            acc = __builtin_amdgcn_wmma_f32_16x16x32_bf16(
                      false, a1, false, h0b[1], (short)0, acc, false, false);

            // lane's 8 results are contiguous in k_out = mt*16 + 8*hf + r
            const int k0 = mt * 16 + 8 * hf;
            v8bf hv;
#pragma unroll
            for (int r = 0; r < 8; ++r)
                hv[r] = (__bf16)fast_celu(acc[r] + s_bh[0][k0 + r]);
            *(v8bf*)(&hbuf[wave][mrow][k0]) = hv;    // one ds_store_b128
        }

        // ---- layer 2 + output fused: same-wave LDS is in-order, no barrier
        const v8bf* hrow = (const v8bf*)(&hbuf[wave][mrow][0]);
        const v16bf h1b0 = cat8(hrow[2 * hf],     hrow[2 * hf + 1]); // K=0..31
        const v16bf h1b1 = cat8(hrow[4 + 2 * hf], hrow[5 + 2 * hf]); // K=32..63

        float part = 0.0f;
#pragma unroll
        for (int mt = 0; mt < 4; ++mt) {
            const v8bf* wrow = (const v8bf*)(&wq[1][mt * 16 + mrow][0]);
            const v16bf a0 = cat8(wrow[2 * hf],     wrow[2 * hf + 1]);
            const v16bf a1 = cat8(wrow[4 + 2 * hf], wrow[5 + 2 * hf]);

            v8f acc = {};
            acc = __builtin_amdgcn_wmma_f32_16x16x32_bf16(
                      false, a0, false, h1b0, (short)0, acc, false, false);
            acc = __builtin_amdgcn_wmma_f32_16x16x32_bf16(
                      false, a1, false, h1b1, (short)0, acc, false, false);

            const int k0 = mt * 16 + 8 * hf;
#pragma unroll
            for (int r = 0; r < 8; ++r)          // output dot folded in (packed fma)
                part += fast_celu(acc[r] + s_bh[1][k0 + r]) * s_wo[k0 + r];
        }
        part += __shfl_xor(part, 16);            // combine the two K halves
        if (hf == 0 && row0 + mrow < Btot)
            out[(size_t)(row0 + mrow) * F + f] = part + s_bo;
    }
}

} // anonymous namespace

extern "C" void kernel_launch(void* const* d_in, const int* in_sizes, int n_in,
                              void* d_out, int out_size, void* d_ws, size_t ws_size,
                              hipStream_t stream) {
    (void)n_in; (void)d_ws; (void)ws_size; (void)out_size;
    const float* x     = (const float*)d_in[0];
    const float* w_in  = (const float*)d_in[1];
    const float* b_in  = (const float*)d_in[2];
    const float* w_hid = (const float*)d_in[3];
    const float* b_hid = (const float*)d_in[4];
    const float* w_out = (const float*)d_in[5];
    const float* b_out = (const float*)d_in[6];
    float* outp = (float*)d_out;

    const int Btot = in_sizes[0] / F;   // x is [B, F]
    dim3 grid(F, (Btot + BATCH_PER_BLOCK - 1) / BATCH_PER_BLOCK);
    dim3 block(WAVES * 32);
    hipLaunchKernelGGL(feature_mlp_kernel, grid, block, 0, stream,
                       x, w_in, b_in, w_hid, b_hid, w_out, b_out, outp, Btot);
}
